// ObliviousDecisionTreeCMB_38637525795488
// MI455X (gfx1250) — matrix operand order repaired and checked by reading, not compiled, not run on hardware
//
#include <hip/hip_runtime.h>
#include <hip/hip_bf16.h>

// Shapes (compile-time constants from the reference)
#define BATCH   512
#define DIM     256
#define NTREES  512
#define DEPTH   6
#define UNITS   3
#define ND      (NTREES * DEPTH)   // 3072
#define NLEAVES 64

typedef __attribute__((ext_vector_type(16))) _Float16 v16h;
typedef __attribute__((ext_vector_type(8)))  _Float16 v8h;
typedef __attribute__((ext_vector_type(8)))  float    v8f;

// ---------------------------------------------------------------------------
// Kernel 0: convert x (f32) -> f16, row-major [b][k]
// ---------------------------------------------------------------------------
__global__ __launch_bounds__(256)
void convert_x_kernel(const float* __restrict__ x, _Float16* __restrict__ Xh) {
    int i = blockIdx.x * 256 + threadIdx.x;   // 0 .. BATCH*DIM-1
    Xh[i] = (_Float16)x[i];
}

// ---------------------------------------------------------------------------
// Kernel 1: sparsemax over feature dim (256) for each (tree,depth) column.
// One 256-thread block per column. Bitonic sort (descending) + inclusive scan
// in LDS, then tau threshold. Output written K-contiguous: ShT[col*256 + k]
// (f16) so the WMMA B-fragment is a single contiguous 32-byte load per lane.
// ---------------------------------------------------------------------------
__global__ __launch_bounds__(256)
void sparsemax_kernel(const float* __restrict__ logits,  // [DIM][ND]
                      _Float16* __restrict__ ShT) {      // [ND][DIM] f16
    const int col = blockIdx.x;     // nd in 0..3071
    const int t   = threadIdx.x;    // 0..255  (feature index)

    __shared__ float s[DIM];   // sorted (descending)
    __shared__ float c[DIM];   // inclusive cumsum of s
    __shared__ int   kz;

    const float z = logits[t * ND + col];
    s[t] = z;
    if (t == 0) kz = 0;
    __syncthreads();

    // Bitonic sort, descending
    for (unsigned k = 2; k <= DIM; k <<= 1) {
        for (unsigned j = k >> 1; j > 0; j >>= 1) {
            unsigned ixj = t ^ j;
            if (ixj > (unsigned)t) {
                float a = s[t], b = s[ixj];
                bool desc = ((t & k) == 0);
                bool swp = desc ? (a < b) : (a > b);
                if (swp) { s[t] = b; s[ixj] = a; }
            }
            __syncthreads();
        }
    }

    // Inclusive scan (Hillis-Steele)
    c[t] = s[t];
    __syncthreads();
    for (int off = 1; off < DIM; off <<= 1) {
        float v = c[t];
        if (t >= off) v += c[t - off];
        __syncthreads();
        c[t] = v;
        __syncthreads();
    }

    // support: (1 + k*zs_k) > cum_k, k = t+1
    int support = (1.0f + (float)(t + 1) * s[t]) > c[t] ? 1 : 0;
    if (support) atomicAdd(&kz, 1);
    __syncthreads();

    const int   k_z = kz;
    const float tau = (c[k_z - 1] - 1.0f) / (float)k_z;
    const float out = fmaxf(z - tau, 0.0f);

    ShT[col * DIM + t] = (_Float16)out;   // coalesced: consecutive t -> consecutive addr
}

// ---------------------------------------------------------------------------
// Kernel 2: fv = x @ selectors via v_wmma_f32_16x16x32_f16, fused epilogue:
//   tl[b][nd] = (fv - thr[nd]) * exp(-logT[nd])
// One wave (32 lanes) per 16x16 output tile; K=256 in 8 WMMA steps.
//
// A fragment (16x32 f16): lane%16 = M row; element e holds
//   K = k0 + (lane/16)*16 + e                -> one contiguous v16h load.
// B fragment (32x16 f16): lane%16 = N col; element e holds
//   K = k0 + (lane/16)*16 + e                -> one contiguous v16h load.
// C/D (16x16 f32): VGPR r, lane L -> M = r + 8*(L/16), N = L%16.
// ---------------------------------------------------------------------------
__global__ __launch_bounds__(256)
void fv_wmma_kernel(const _Float16* __restrict__ Xh,    // [BATCH][DIM] f16
                    const _Float16* __restrict__ ShT,   // [ND][DIM]   f16
                    const float* __restrict__ thr,      // [ND]
                    const float* __restrict__ logT,     // [ND]
                    float* __restrict__ TL) {           // [BATCH][ND] f32
    const int lane = threadIdx.x & 31;      // gfx1250 is wave32
    const int wave = threadIdx.x >> 5;
    const int tilesN = ND / 16;             // 192
    const int tile = blockIdx.x * 8 + wave; // 8 waves per block
    const int m0 = (tile / tilesN) * 16;
    const int n0 = (tile % tilesN) * 16;

    const int lm = lane & 15;   // M (for A) / N (for B)
    const int g  = lane >> 4;   // lane half-group

    const _Float16* __restrict__ arow = Xh  + (m0 + lm) * DIM;
    const _Float16* __restrict__ brow = ShT + (n0 + lm) * DIM;

    v8f acc = {};
#pragma unroll
    for (int k0 = 0; k0 < DIM; k0 += 32) {
        v16h a = *(const v16h*)(arow + k0 + g * 16);

        v16h b = *(const v16h*)(brow + k0 + g * 16);

        acc = __builtin_amdgcn_wmma_f32_16x16x32_f16(
            /*neg_a=*/false, a, /*neg_b=*/false, b,
            /*c_mod=*/(short)0, acc, /*reuse_a=*/false, /*reuse_b=*/false);
    }

    // Fused epilogue: tl = (fv - thr) * exp(-logT)
    const int   nd  = n0 + lm;
    const float t0  = thr[nd];
    const float itp = __expf(-logT[nd]);
#pragma unroll
    for (int r = 0; r < 8; ++r) {
        const int brow_i = m0 + r + g * 8;          // batch row for acc[r]
        TL[brow_i * ND + nd] = (acc[r] - t0) * itp;
    }
}

// ---------------------------------------------------------------------------
// Kernel 3: gates -> leaf products -> response.
// One thread per (b, n): read 6 tl values, form per-level gates,
// product over 64 leaves, dot with response[n, u, :].
// block = 512 threads over b (response[n] broadcast within block), grid = n.
// ---------------------------------------------------------------------------
__global__ __launch_bounds__(512)
void leaf_kernel(const float* __restrict__ TL,        // [BATCH][ND]
                 const float* __restrict__ response,  // [NTREES][UNITS][NLEAVES]
                 float* __restrict__ out) {           // [BATCH][NTREES][UNITS]
    const int n = blockIdx.x;    // tree
    const int b = threadIdx.x;   // batch

    float g0[DEPTH], g1[DEPTH];
#pragma unroll
    for (int d = 0; d < DEPTH; ++d) {
        float tl = TL[b * ND + n * DEPTH + d];
        g1[d] = fminf(fmaxf(0.5f * tl + 0.5f, 0.0f), 1.0f);  // sparsemoid(tl)
        g0[d] = fminf(fmaxf(0.5f - 0.5f * tl, 0.0f), 1.0f);  // sparsemoid(-tl)
    }

    const float* __restrict__ resp = response + n * UNITS * NLEAVES;
    float a0 = 0.0f, a1 = 0.0f, a2 = 0.0f;
#pragma unroll 4
    for (int c = 0; c < NLEAVES; ++c) {
        float p = 1.0f;
#pragma unroll
        for (int d = 0; d < DEPTH; ++d)
            p *= ((c >> d) & 1) ? g0[d] : g1[d];   // lut: bit=1 -> sparsemoid(-tl)
        a0 = fmaf(p, resp[c], a0);
        a1 = fmaf(p, resp[NLEAVES + c], a1);
        a2 = fmaf(p, resp[2 * NLEAVES + c], a2);
    }

    float* o = out + (b * NTREES + n) * UNITS;
    o[0] = a0; o[1] = a1; o[2] = a2;
}

// ---------------------------------------------------------------------------
// Launch
// ---------------------------------------------------------------------------
extern "C" void kernel_launch(void* const* d_in, const int* in_sizes, int n_in,
                              void* d_out, int out_size, void* d_ws, size_t ws_size,
                              hipStream_t stream) {
    const float* x      = (const float*)d_in[0];  // [BATCH][DIM]
    const float* logits = (const float*)d_in[1];  // [DIM][NTREES][DEPTH]
    const float* thr    = (const float*)d_in[2];  // [NTREES][DEPTH]
    const float* logT   = (const float*)d_in[3];  // [NTREES][DEPTH]
    const float* resp   = (const float*)d_in[4];  // [NTREES][UNITS][NLEAVES]
    float* out          = (float*)d_out;          // [BATCH][NTREES][UNITS]

    // Workspace layout
    char* ws = (char*)d_ws;
    _Float16* Xh  = (_Float16*)(ws);                                   // 256 KB
    _Float16* ShT = (_Float16*)(ws + (size_t)BATCH * DIM * 2);         // 1.5 MB
    float*    TL  = (float*)(ws + (size_t)BATCH * DIM * 2
                                + (size_t)ND * DIM * 2);               // 6 MB

    convert_x_kernel<<<(BATCH * DIM) / 256, 256, 0, stream>>>(x, Xh);
    sparsemax_kernel<<<ND, DIM, 0, stream>>>(logits, ShT);

    const int nTiles = (BATCH / 16) * (ND / 16);   // 32 * 192 = 6144 tiles
    fv_wmma_kernel<<<nTiles / 8, 256, 0, stream>>>(Xh, ShT, thr, logT, TL);

    leaf_kernel<<<NTREES, BATCH, 0, stream>>>(TL, resp, out);
}
